// AttentionKernelIntegral_85564338471214
// MI455X (gfx1250) — matrix operand order repaired and checked by reading, not compiled, not run on hardware
//
#include <hip/hip_runtime.h>
#include <hip/hip_bf16.h>
#include <cstdint>
#include <cstddef>

// ---------------------------------------------------------------------------
// MI455X / gfx1250, wave32. All matrix math via V_WMMA_F32_16X16X32_F16.
// Fragment layouts per CDNA5 ISA 7.12.2:
//   A (16x32 f16): lane l: M = l&15; halves 0..7 = K koff..koff+7,
//                  halves 8..15 = K koff+16..koff+23, koff = (l>=16)*8
//   B (32x16 f16): lane l: N = l&15; ISA K order as A; code loads A, B as halves 0..15 = K koff2..koff2+15,
//                  koff2 = (l>=16)*16
//   C/D (16x16 f32, v8f): element r: M = r + 8*(l>=16), N = l&15
// ---------------------------------------------------------------------------

typedef __attribute__((ext_vector_type(16))) _Float16 v16h;
typedef __attribute__((ext_vector_type(8)))  _Float16 v8h;
typedef __attribute__((ext_vector_type(4)))  _Float16 v4h;
typedef __attribute__((ext_vector_type(8)))  float    v8f;

#define NGRID 16384
#define BATCH 4
#define CIN   256
#define HEADS 8
#define DHEAD 64
#define HD    512              // HEADS*DHEAD
#define QKVW  1536             // 3*HD
#define MROWS (BATCH * NGRID)  // 65536
#define OUTC  256

static __device__ __forceinline__ v16h cat8(v8h lo, v8h hi) {
  v16h r;
#pragma unroll
  for (int i = 0; i < 8; ++i) { r[i] = lo[i]; r[i + 8] = hi[i]; }
  return r;
}

static __device__ __forceinline__ v8f wmma_f16(v16h a, v16h b, v8f c) {
  return __builtin_amdgcn_wmma_f32_16x16x32_f16(false, a, false, b, (short)0, c,
                                                false, false);
}

// ---------------------------------------------------------------------------
// f32 -> f16 cast (grid-stride)
// ---------------------------------------------------------------------------
__global__ void cast_f32_to_f16_kernel(const float* __restrict__ src,
                                       _Float16* __restrict__ dst, int n) {
  int i = blockIdx.x * blockDim.x + threadIdx.x;
  int stride = gridDim.x * blockDim.x;
  for (; i < n; i += stride) dst[i] = (_Float16)src[i];
}

// ---------------------------------------------------------------------------
// C[M x Ntot] = A[M x K] * W[Ntot x K]^T  (+ bias), f16 inputs, f32 accum.
// 256 threads = 8 waves, block tile 128x128, wave tile 32x64 (2x4 WMMA tiles).
// ---------------------------------------------------------------------------
template <bool OUT_F16, bool BIAS>
__global__ void gemm_awt_kernel(const _Float16* __restrict__ A,
                                const _Float16* __restrict__ W,
                                const float* __restrict__ bias,
                                _Float16* __restrict__ Of16,
                                float* __restrict__ Of32, int K, int Ntot) {
  const int lane = threadIdx.x & 31;
  const int wave = threadIdx.x >> 5;
  const int wm = wave >> 1;  // 0..3
  const int wn = wave & 1;   // 0..1
  const int mbase = blockIdx.x * 128 + wm * 32;
  const int nbase = blockIdx.y * 128 + wn * 64;
  const int ml = lane & 15;
  const int hl = lane >> 4;

  v8f acc[2][4] = {};
  for (int k0 = 0; k0 < K; k0 += 32) {
    v16h af[2], bf[4];
#pragma unroll
    for (int a = 0; a < 2; ++a) {
      const _Float16* p = A + (size_t)(mbase + a * 16 + ml) * K + (k0 + hl * 16);
      af[a] = *(const v16h*)p;
    }
#pragma unroll
    for (int f = 0; f < 4; ++f) {
      const _Float16* p = W + (size_t)(nbase + f * 16 + ml) * K + (k0 + hl * 16);
      bf[f] = *(const v16h*)p;
    }
#pragma unroll
    for (int a = 0; a < 2; ++a)
#pragma unroll
      for (int f = 0; f < 4; ++f) acc[a][f] = wmma_f16(af[a], bf[f], acc[a][f]);
  }

#pragma unroll
  for (int a = 0; a < 2; ++a) {
#pragma unroll
    for (int f = 0; f < 4; ++f) {
      const int row0 = mbase + a * 16 + hl * 8;
      const int col = nbase + f * 16 + ml;
      float bv = 0.0f;
      if constexpr (BIAS) bv = bias[col];
#pragma unroll
      for (int r = 0; r < 8; ++r) {
        const float v = acc[a][f][r] + bv;
        const size_t off = (size_t)(row0 + r) * Ntot + col;
        if constexpr (OUT_F16) Of16[off] = (_Float16)v;
        else Of32[off] = v;
      }
    }
  }
}

// ---------------------------------------------------------------------------
// Instance-norm stats: one block per (b, h, which in {k,v}); 512 threads.
// stats layout: [(b*H+h)*256 + which*128 + {0:mean,1:rstd}*64 + d]
// ---------------------------------------------------------------------------
__global__ void stats_kernel(const _Float16* __restrict__ qkv,
                             float* __restrict__ stats) {
  const int bid = blockIdx.x;
  const int which = bid & 1;
  const int h = (bid >> 1) & 7;
  const int b = bid >> 4;
  const int colbase = HD + which * HD + h * DHEAD;
  const int t = threadIdx.x;
  const int d = t & 63;
  const int nn = t >> 6;  // 0..7

  float s = 0.f, sq = 0.f;
  for (int n = nn; n < NGRID; n += 8) {
    float x = (float)qkv[(size_t)(b * NGRID + n) * QKVW + colbase + d];
    s += x;
    sq += x * x;
  }
  __shared__ float rs[512], rq[512];
  rs[t] = s;
  rq[t] = sq;
  __syncthreads();
  if (t < 64) {
    float ts = 0.f, tq = 0.f;
#pragma unroll
    for (int j = 0; j < 8; ++j) { ts += rs[j * 64 + d]; tq += rq[j * 64 + d]; }
    const float mean = ts * (1.0f / NGRID);
    const float var = tq * (1.0f / NGRID) - mean * mean;
    const float rstd = rsqrtf(var + 1e-5f);
    const int base = ((b * HEADS + h) * 2 + which) * 128;
    stats[base + d] = mean;
    stats[base + 64 + d] = rstd;
  }
}

// ---------------------------------------------------------------------------
// kv[b,h] = (1/N) * k_norm^T @ v_norm  (64x64 per (b,h), K = NGRID).
// 512 threads = 16 waves; each wave owns one 16x16 output tile.
// k,v tiles staged TRANSPOSED in LDS ([d][n]) with instance-norm applied,
// so both A and B WMMA fragments are contiguous ds loads.
// ---------------------------------------------------------------------------
__global__ void kv_kernel(const _Float16* __restrict__ qkv,
                          const float* __restrict__ stats,
                          _Float16* __restrict__ kvout) {
  __shared__ alignas(32) _Float16 kT[DHEAD * 32];
  __shared__ alignas(32) _Float16 vT[DHEAD * 32];
  __shared__ float sm[256];

  const int bh = blockIdx.x;
  const int b = bh >> 3;
  const int h = bh & 7;
  const int t = threadIdx.x;  // 0..511
  if (t < 256) sm[t] = stats[(size_t)bh * 256 + t];

  const int lane = t & 31;
  const int wave = t >> 5;
  const int d0 = (wave >> 2) * 16;
  const int e0 = (wave & 3) * 16;
  const int ml = lane & 15;
  const int hl = lane >> 4;
  const int nn = t >> 4;         // 0..31 (row within k-tile)
  const int dq = (t & 15) * 4;   // 0..60 (4 channels per thread)

  v8f acc = {};
  __syncthreads();  // sm ready

  for (int n0 = 0; n0 < NGRID; n0 += 32) {
    const size_t rbase = (size_t)(b * NGRID + n0 + nn) * QKVW;
    const v4h kk = *(const v4h*)(qkv + rbase + HD + h * DHEAD + dq);
    const v4h vv = *(const v4h*)(qkv + rbase + 2 * HD + h * DHEAD + dq);
    __syncthreads();  // previous tile fully consumed
#pragma unroll
    for (int j = 0; j < 4; ++j) {
      const int dd = dq + j;
      kT[dd * 32 + nn] = (_Float16)(((float)kk[j] - sm[dd]) * sm[64 + dd]);
      vT[dd * 32 + nn] = (_Float16)(((float)vv[j] - sm[128 + dd]) * sm[192 + dd]);
    }
    __syncthreads();
    const _Float16* ap = &kT[(d0 + ml) * 32 + hl * 16];
    const v16h af = *(const v16h*)ap;
    const v16h bf = *(const v16h*)&vT[(e0 + ml) * 32 + hl * 16];
    acc = wmma_f16(af, bf, acc);
  }

  const float scale = 1.0f / NGRID;
#pragma unroll
  for (int r = 0; r < 8; ++r) {
    const int row = d0 + r + hl * 8;
    const int col = e0 + ml;
    kvout[(size_t)bh * 4096 + row * 64 + col] = (_Float16)(acc[r] * scale);
  }
}

// ---------------------------------------------------------------------------
// attn[:, h*64:(h+1)*64] = q[:, h*64:(h+1)*64] @ kv[b,h]   (K = 64)
// 256 threads = 8 waves; wave tile 16x64; kv staged transposed in LDS.
// ---------------------------------------------------------------------------
__global__ void attn_kernel(const _Float16* __restrict__ qkv,
                            const _Float16* __restrict__ kv,
                            _Float16* __restrict__ attn) {
  __shared__ alignas(32) _Float16 kvT[DHEAD * DHEAD];
  const int h = blockIdx.y;
  const int mbase = blockIdx.x * 128;
  const int b = mbase / NGRID;
  const int t = threadIdx.x;
  {
    const _Float16* src = kv + (size_t)(b * HEADS + h) * 4096;
    const int din = t >> 2;
    const int e0 = (t & 3) * 16;
    const v16h row = *(const v16h*)(src + din * 64 + e0);
#pragma unroll
    for (int j = 0; j < 16; ++j) kvT[(e0 + j) * 64 + din] = row[j];
  }
  __syncthreads();

  const int lane = t & 31;
  const int wave = t >> 5;
  const int mrow = mbase + wave * 16;
  const int ml = lane & 15;
  const int hl = lane >> 4;

  v8f acc[4] = {};
#pragma unroll
  for (int k0 = 0; k0 < 64; k0 += 32) {
    const _Float16* p = qkv + (size_t)(mrow + ml) * QKVW + h * DHEAD + k0 + hl * 16;
    const v16h af = *(const v16h*)p;
#pragma unroll
    for (int f = 0; f < 4; ++f) {
      const v16h bf = *(const v16h*)&kvT[(f * 16 + ml) * 64 + k0 + hl * 16];
      acc[f] = wmma_f16(af, bf, acc[f]);
    }
  }
#pragma unroll
  for (int f = 0; f < 4; ++f) {
    const int row0 = mrow + hl * 8;
    const int col = h * DHEAD + f * 16 + ml;
#pragma unroll
    for (int r = 0; r < 8; ++r)
      attn[(size_t)(row0 + r) * HD + col] = (_Float16)acc[f][r];
  }
}

// ---------------------------------------------------------------------------
extern "C" void kernel_launch(void* const* d_in, const int* in_sizes, int n_in,
                              void* d_out, int out_size, void* d_ws,
                              size_t ws_size, hipStream_t stream) {
  (void)in_sizes; (void)n_in; (void)out_size; (void)ws_size;
  const float* u_src = (const float*)d_in[0];
  // d_in[1] = pos_src (unused by reference math)
  const float* Wq = (const float*)d_in[2];
  const float* Wk = (const float*)d_in[3];
  const float* Wv = (const float*)d_in[4];
  const float* Wo = (const float*)d_in[5];
  const float* bo = (const float*)d_in[6];
  float* out = (float*)d_out;

  char* ws = (char*)d_ws;
  size_t off = 0;
  auto alloc = [&](size_t bytes) -> void* {
    void* p = ws + off;
    off = (off + bytes + 255) & ~(size_t)255;
    return p;
  };
  _Float16* u16   = (_Float16*)alloc((size_t)MROWS * CIN * 2);
  _Float16* w16   = (_Float16*)alloc((size_t)QKVW * CIN * 2);
  _Float16* wo16  = (_Float16*)alloc((size_t)OUTC * HD * 2);
  float*    stats = (float*)alloc((size_t)BATCH * HEADS * 256 * 4);
  _Float16* qkv16 = (_Float16*)alloc((size_t)MROWS * QKVW * 2);
  _Float16* kv16  = (_Float16*)alloc((size_t)BATCH * HEADS * DHEAD * DHEAD * 2);
  _Float16* at16  = (_Float16*)alloc((size_t)MROWS * HD * 2);

  // 1) casts
  cast_f32_to_f16_kernel<<<4096, 256, 0, stream>>>(u_src, u16, MROWS * CIN);
  cast_f32_to_f16_kernel<<<128, 256, 0, stream>>>(Wq, w16, HD * CIN);
  cast_f32_to_f16_kernel<<<128, 256, 0, stream>>>(Wk, w16 + HD * CIN, HD * CIN);
  cast_f32_to_f16_kernel<<<128, 256, 0, stream>>>(Wv, w16 + 2 * HD * CIN, HD * CIN);
  cast_f32_to_f16_kernel<<<128, 256, 0, stream>>>(Wo, wo16, OUTC * HD);

  // 2) fused QKV projection: [65536 x 256] x [1536 x 256]^T -> f16
  gemm_awt_kernel<true, false><<<dim3(MROWS / 128, QKVW / 128), 256, 0, stream>>>(
      u16, w16, nullptr, qkv16, nullptr, CIN, QKVW);

  // 3) instance-norm stats for k and v
  stats_kernel<<<BATCH * HEADS * 2, 512, 0, stream>>>(qkv16, stats);

  // 4) kv = (1/N) * k_norm^T @ v_norm per (b,h)
  kv_kernel<<<BATCH * HEADS, 512, 0, stream>>>(qkv16, stats, kv16);

  // 5) u2 = q @ kv (block-diagonal per head)
  attn_kernel<<<dim3(MROWS / 128, HEADS), 256, 0, stream>>>(qkv16, kv16, at16);

  // 6) out = u2 @ Wo^T + bo -> f32 d_out
  gemm_awt_kernel<false, true><<<dim3(MROWS / 128, OUTC / 128), 256, 0, stream>>>(
      at16, wo16, bo, nullptr, out, HD, OUTC);
}
